// SSMConv1D_20444044329275
// MI455X (gfx1250) — matrix operand order repaired and checked by reading, not compiled, not run on hardware
//
#include <hip/hip_runtime.h>
#include <hip/hip_bf16.h>

// ---------------------------------------------------------------------------
// Mamba block (DEPTH=8) for MI455X / gfx1250, wave32, WMMA bf16 matrix path,
// async global->LDS staging (ASYNCcnt) for B tiles.
// ---------------------------------------------------------------------------
#define DEPTH  8
#define DIM    1024
#define DI     2048
#define DSTATE 16
#define DCONV  4
#define DTRANK 64
#define BB     2
#define LL     2048
#define MROWS  (BB * LL)    // 4096
#define CHUNK  64           // scan chunks along L
#define CLEN   (LL / CHUNK) // 32 steps per chunk

// GEMM tiling
#define BN      128               // block N tile
#define BK      32                // k step
#define LDS_ROW 40                // ushorts per LDS row (32 data + 8 pad -> 20 dwords,
                                  // r*20 mod 64 is a permutation: conflict-free b128 reads)
#define LDS_BUF (BN * LDS_ROW)    // 5120 ushorts = 10 KB per buffer

#if defined(__has_builtin)
#if __has_builtin(__builtin_amdgcn_global_load_async_to_lds_b128)
#define HAVE_ASYNC_LDS 1
#endif
#endif
#ifndef HAVE_ASYNC_LDS
#define HAVE_ASYNC_LDS 0
#endif

typedef __attribute__((ext_vector_type(16))) __bf16 v16bf;
typedef __attribute__((ext_vector_type(8)))  float  v8f;
typedef int vec4i __attribute__((vector_size(16)));
typedef __attribute__((address_space(1))) vec4i* gv4p;   // global int4*
typedef __attribute__((address_space(3))) vec4i* lv4p;   // LDS int4*

__device__ __forceinline__ unsigned short f2bf(float f) {
    unsigned int u = __builtin_bit_cast(unsigned int, f);
    u += 0x7fffu + ((u >> 16) & 1u);               // round-to-nearest-even
    return (unsigned short)(u >> 16);
}
__device__ __forceinline__ float bf2f(unsigned short h) {
    return __builtin_bit_cast(float, (unsigned int)h << 16);
}

__device__ __forceinline__ void async_wait_all() {
#if HAVE_ASYNC_LDS
#if __has_builtin(__builtin_amdgcn_s_wait_asynccnt)
    __builtin_amdgcn_s_wait_asynccnt(0);
#else
    asm volatile("s_wait_asynccnt 0x0" ::: "memory");
#endif
#endif
}

// ---------------------------------------------------------------------------
// f32 -> bf16 bulk convert (weights + activations), grid-stride.
// ---------------------------------------------------------------------------
__global__ void k_cvt(const float* __restrict__ src, unsigned short* __restrict__ dst,
                      size_t n) {
    size_t i = (size_t)blockIdx.x * blockDim.x + threadIdx.x;
    size_t stride = (size_t)gridDim.x * blockDim.x;
    for (; i < n; i += stride) dst[i] = f2bf(src[i]);
}

// ---------------------------------------------------------------------------
// WMMA GEMM:  C[M,N] = act( A[M,K] * W[N,K]^T + bias[N] )
// A, W bf16 (K contiguous). 256 threads = 8 waves arranged 4(M) x 2(N);
// wave tile 32x64 = 2x4 v_wmma_f32_16x16x32_bf16 tiles; block tile 128x128.
// B tile (128x32) staged ONCE per block into LDS via async global->LDS copy
// (double-buffered, overlapped with WMMAs); A fragments direct from global
// with global_prefetch of the next k-tile.
// ACT: 0 = none, 1 = bias add + softplus.
// ---------------------------------------------------------------------------
template <int ACT, typename OUT>
__global__ __launch_bounds__(256) void k_gemm(
    const unsigned short* __restrict__ A, int lda,
    const unsigned short* __restrict__ W, int ldw,
    OUT* __restrict__ C, int ldc,
    int M, int N, int K,
    const float* __restrict__ bias)
{
    __shared__ unsigned short smem[2 * LDS_BUF];   // 20 KB of the 320 KB LDS

    const int tid  = threadIdx.x;
    const int lane = tid & 31;
    const int wave = tid >> 5;
    const int wm = wave >> 1;          // 0..3
    const int wn = wave & 1;           // 0..1
    const int gm  = blockIdx.y * 128 + wm * 32;
    const int gnb = blockIdx.x * BN;   // block column base
    const int wnb = wn * 64;           // wave column base within block
    const int r  = lane & 15;
    const int hh = lane >> 4;          // half-wave select

    // Stage one 128x32 bf16 B tile into LDS buffer `buf` (512 x 16B chunks,
    // 2 chunks per thread) using the async copy engine when available.
    auto stage = [&](int buf, int k0) {
#pragma unroll
        for (int c = tid; c < BN * 4; c += 256) {
            int row = c >> 2, sub = c & 3;
            int n = gnb + row; if (n > N - 1) n = N - 1;          // edge clamp
            const unsigned short* gp = W + (size_t)n * ldw + k0 + sub * 8;
            unsigned short* lp = &smem[buf * LDS_BUF + row * LDS_ROW + sub * 8];
#if HAVE_ASYNC_LDS
            __builtin_amdgcn_global_load_async_to_lds_b128((gv4p)gp, (lv4p)lp, 0, 0);
#else
            *(uint4*)lp = *(const uint4*)gp;
#endif
        }
    };

    v8f acc[2][4] = {};

    stage(0, 0);
    async_wait_all();
    __syncthreads();

    int buf = 0;
    for (int k0 = 0; k0 < K; k0 += BK) {
        const int kn = k0 + BK;
        if (kn < K) {
            stage(buf ^ 1, kn);                     // overlap with compute
            __builtin_prefetch(A + (size_t)(gm + r) * lda + kn, 0, 1);
        }

        v16bf a[2], b[4];
        // A fragment (global): lane = row r; elems0-7 <- K=k0+hh*8..+7,
        //                                    elems8-15 <- K=k0+hh*8+16..+23
#pragma unroll
        for (int t = 0; t < 2; ++t) {
            const unsigned short* p = A + (size_t)(gm + t * 16 + r) * lda + k0 + hh * 8;
            union { uint4 q[2]; v16bf v; } f;
            f.q[0] = *(const uint4*)(p);
            f.q[1] = *(const uint4*)(p + 16);
            a[t] = f.v;
        }
        // B fragment (LDS): lane = column; elems0-7 <- K=hh*8..+7, elems8-15 <- K=hh*8+16..+23
#pragma unroll
        for (int t = 0; t < 4; ++t) {
            const unsigned short* p =
                &smem[buf * LDS_BUF + (wnb + t * 16 + r) * LDS_ROW + hh * 8];
            union { uint4 q[2]; v16bf v; } f;
            f.q[0] = *(const uint4*)(p);
            f.q[1] = *(const uint4*)(p + 16);
            b[t] = f.v;
        }
#pragma unroll
        for (int i = 0; i < 2; ++i)
#pragma unroll
            for (int j = 0; j < 4; ++j)
                acc[i][j] = __builtin_amdgcn_wmma_f32_16x16x32_bf16(
                    false, a[i], false, b[j], (short)0, acc[i][j], false, false);

        if (kn < K) {
            async_wait_all();        // own async stores to buf^1 complete
            __syncthreads();         // => all waves' stores visible; reads of buf done
            buf ^= 1;
        }
    }

    // Store: C/D layout: col = lane&15, row = p + 8*(lane>=16)
    const int col = lane & 15;
    const int rowoff = hh * 8;
#pragma unroll
    for (int i = 0; i < 2; ++i)
#pragma unroll
        for (int j = 0; j < 4; ++j) {
            int n = gnb + wnb + j * 16 + col;
            if (n >= N) continue;
#pragma unroll
            for (int p = 0; p < 8; ++p) {
                int m = gm + i * 16 + rowoff + p;
                float v = acc[i][j][p];
                if constexpr (ACT == 1) {
                    v += bias[n];
                    v = (v > 20.f) ? v : log1pf(__expf(v));
                }
                if constexpr (sizeof(OUT) == 2)
                    ((unsigned short*)C)[(size_t)m * ldc + n] = f2bf(v);
                else
                    ((float*)C)[(size_t)m * ldc + n] = v;
            }
        }
}

// ---------------------------------------------------------------------------
// Depthwise causal conv1d (width 4) + SiLU.  xz: [M, 2*DI] bf16 (xs in cols
// 0..DI-1).  Output xs_act: [M, DI] bf16.
// ---------------------------------------------------------------------------
__global__ __launch_bounds__(256) void k_conv(
    const unsigned short* __restrict__ xz,
    const float* __restrict__ cw, const float* __restrict__ cb,
    unsigned short* __restrict__ xs)
{
    int idx = blockIdx.x * blockDim.x + threadIdx.x;   // over M*DI = 2^23
    int d  = idx & (DI - 1);
    int bl = idx >> 11;          // DI = 2^11
    int b  = bl >> 11;           // LL = 2^11
    int l  = bl & (LL - 1);
    float s = cb[d];
#pragma unroll
    for (int j = 0; j < DCONV; ++j) {
        int ls = l - (DCONV - 1) + j;
        if (ls >= 0)
            s += cw[d * DCONV + j] * bf2f(xz[((size_t)(b * LL + ls)) * (2 * DI) + d]);
    }
    float act = s / (1.f + __expf(-s));                // SiLU
    xs[(size_t)bl * DI + d] = f2bf(act);
}

// ---------------------------------------------------------------------------
// Chunk-parallel selective scan (3 passes). Per-state chunk transfer is
// prod_t exp(dt_t*a_n) = exp(a_n * sum_t dt_t), so chunk summaries are just
// (local final h[16], sum dt): 64 chunks x 4096 channels = 262k-way parallel.
// ---------------------------------------------------------------------------
__global__ __launch_bounds__(256) void k_scan1(
    const unsigned short* __restrict__ xs,     // [M, DI] bf16
    const float*          __restrict__ delta,  // [M, DI] f32
    const unsigned short* __restrict__ xdbl,   // [M, 96] bf16 (B @64)
    const float*          __restrict__ A_log,  // [DI, 16]
    float* __restrict__ hsc,                   // [B, CHUNK, 16, DI]
    float* __restrict__ sdt)                   // [B, CHUNK, DI]
{
    const int d     = (blockIdx.x & 7) * 256 + threadIdx.x;
    const int chunk = (blockIdx.x >> 3) & (CHUNK - 1);
    const int b     = blockIdx.x >> 9;
    float a[DSTATE], h[DSTATE];
#pragma unroll
    for (int n = 0; n < DSTATE; ++n) {
        a[n] = -__expf(A_log[(size_t)d * DSTATE + n]);
        h[n] = 0.f;
    }
    float S = 0.f;
    const int l0 = chunk * CLEN;
    for (int t = 0; t < CLEN; ++t) {
        size_t row = (size_t)b * LL + l0 + t;
        float dt = delta[row * DI + d];
        float xv = bf2f(xs[row * DI + d]);
        float dx = dt * xv;
        S += dt;
#pragma unroll
        for (int n = 0; n < DSTATE; ++n) {
            float bn = bf2f(xdbl[row * 96 + DTRANK + n]);
            h[n] = __expf(dt * a[n]) * h[n] + dx * bn;
        }
    }
    size_t base = ((size_t)(b * CHUNK + chunk) * DSTATE) * DI + d;
#pragma unroll
    for (int n = 0; n < DSTATE; ++n) hsc[base + (size_t)n * DI] = h[n];
    sdt[(size_t)(b * CHUNK + chunk) * DI + d] = S;
}

// Pass 2: sequential combine over the 64 chunk summaries per (b,d);
// rewrite hsc[b,c,:,d] with the state ENTERING chunk c.
__global__ __launch_bounds__(256) void k_scan2(
    const float* __restrict__ A_log,
    float* __restrict__ hsc, const float* __restrict__ sdt)
{
    const int d = (blockIdx.x & 7) * 256 + threadIdx.x;
    const int b = blockIdx.x >> 3;                     // grid = B*8
    float a[DSTATE], h[DSTATE];
#pragma unroll
    for (int n = 0; n < DSTATE; ++n) {
        a[n] = -__expf(A_log[(size_t)d * DSTATE + n]);
        h[n] = 0.f;
    }
    for (int c = 0; c < CHUNK; ++c) {
        size_t base = ((size_t)(b * CHUNK + c) * DSTATE) * DI + d;
        float S = sdt[(size_t)(b * CHUNK + c) * DI + d];
#pragma unroll
        for (int n = 0; n < DSTATE; ++n) {
            float hloc = hsc[base + (size_t)n * DI];
            hsc[base + (size_t)n * DI] = h[n];         // incoming state
            h[n] = hloc + __expf(a[n] * S) * h[n];
        }
    }
}

// Pass 3: re-scan each chunk from its incoming state; emit
// y = (scan + xs*D) * silu(z).
__global__ __launch_bounds__(256) void k_scan3(
    const unsigned short* __restrict__ xs,
    const float*          __restrict__ delta,
    const unsigned short* __restrict__ xdbl,   // B @64, C @80
    const unsigned short* __restrict__ xz,     // z @ col DI
    const float*          __restrict__ A_log,
    const float*          __restrict__ Dp,
    const float*          __restrict__ hsc,
    unsigned short*       __restrict__ y)
{
    const int d     = (blockIdx.x & 7) * 256 + threadIdx.x;
    const int chunk = (blockIdx.x >> 3) & (CHUNK - 1);
    const int b     = blockIdx.x >> 9;
    float a[DSTATE], h[DSTATE];
    size_t base = ((size_t)(b * CHUNK + chunk) * DSTATE) * DI + d;
#pragma unroll
    for (int n = 0; n < DSTATE; ++n) {
        a[n] = -__expf(A_log[(size_t)d * DSTATE + n]);
        h[n] = hsc[base + (size_t)n * DI];
    }
    const float dpar = Dp[d];
    const int l0 = chunk * CLEN;
    for (int t = 0; t < CLEN; ++t) {
        size_t row = (size_t)b * LL + l0 + t;
        float dt = delta[row * DI + d];
        float xv = bf2f(xs[row * DI + d]);
        float zv = bf2f(xz[row * (2 * DI) + DI + d]);
        float dx = dt * xv;
        float yv = 0.f;
#pragma unroll
        for (int n = 0; n < DSTATE; ++n) {
            float bn = bf2f(xdbl[row * 96 + DTRANK + n]);
            float cn = bf2f(xdbl[row * 96 + DTRANK + DSTATE + n]);
            h[n] = __expf(dt * a[n]) * h[n] + dx * bn;
            yv  += h[n] * cn;
        }
        yv += xv * dpar;
        yv *= zv / (1.f + __expf(-zv));                // * silu(z)
        y[row * DI + d] = f2bf(yv);
    }
}

// ---------------------------------------------------------------------------
// RMSNorm over last dim (1024). One block (256 thr = 8 waves) per row.
// ---------------------------------------------------------------------------
template <typename OUT>
__global__ __launch_bounds__(256) void k_rms(
    const float* __restrict__ x, const float* __restrict__ w, OUT* __restrict__ out)
{
    __shared__ float red[8];
    const int row = blockIdx.x;
    const float* xr = x + (size_t)row * DIM;
    float s = 0.f;
    for (int c = threadIdx.x; c < DIM; c += 256) { float v = xr[c]; s += v * v; }
#pragma unroll
    for (int off = 16; off > 0; off >>= 1) s += __shfl_xor(s, off, 32);
    if ((threadIdx.x & 31) == 0) red[threadIdx.x >> 5] = s;
    __syncthreads();
    if (threadIdx.x < 8) {
        float t = red[threadIdx.x];
#pragma unroll
        for (int off = 4; off > 0; off >>= 1) t += __shfl_xor(t, off, 8);
        if (threadIdx.x == 0) red[0] = t;
    }
    __syncthreads();
    const float scale = rsqrtf(red[0] * (1.f / DIM) + 1e-5f);
    for (int c = threadIdx.x; c < DIM; c += 256) {
        float v = xr[c] * scale * w[c];
        if constexpr (sizeof(OUT) == 2)
            ((unsigned short*)out)[(size_t)row * DIM + c] = f2bf(v);
        else
            ((float*)out)[(size_t)row * DIM + c] = v;
    }
}

// ---------------------------------------------------------------------------
// Host orchestration
// ---------------------------------------------------------------------------
extern "C" void kernel_launch(void* const* d_in, const int* in_sizes, int n_in,
                              void* d_out, int out_size, void* d_ws, size_t ws_size,
                              hipStream_t stream) {
    (void)in_sizes; (void)n_in; (void)out_size; (void)ws_size;

    const float* x_in      = (const float*)d_in[0];
    const float* in_proj_w = (const float*)d_in[1];
    const float* conv_w    = (const float*)d_in[2];
    const float* conv_b    = (const float*)d_in[3];
    const float* x_proj_w  = (const float*)d_in[4];
    const float* dt_proj_w = (const float*)d_in[5];
    const float* dt_proj_b = (const float*)d_in[6];
    const float* A_log     = (const float*)d_in[7];
    const float* Dp        = (const float*)d_in[8];
    const float* out_proj_w= (const float*)d_in[9];
    const float* rms_w     = (const float*)d_in[10];

    // ---- workspace layout (bytes, 256-aligned), total ~250 MB ----
    char* ws = (char*)d_ws;
    size_t off = 0;
    auto alloc = [&](size_t bytes) { char* p = ws + off; off += (bytes + 255) & ~(size_t)255; return p; };

    unsigned short* wbf_in  = (unsigned short*)alloc((size_t)DEPTH * 2 * DI * DIM * 2);
    unsigned short* wbf_xp  = (unsigned short*)alloc((size_t)DEPTH * 96 * DI * 2);
    unsigned short* wbf_dt  = (unsigned short*)alloc((size_t)DEPTH * DI * DTRANK * 2);
    unsigned short* wbf_op  = (unsigned short*)alloc((size_t)DEPTH * DIM * DI * 2);
    unsigned short* buf_x   = (unsigned short*)alloc((size_t)MROWS * DIM * 2);
    unsigned short* buf_xz  = (unsigned short*)alloc((size_t)MROWS * 2 * DI * 2);
    unsigned short* buf_xs  = (unsigned short*)alloc((size_t)MROWS * DI * 2);
    unsigned short* buf_xdbl= (unsigned short*)alloc((size_t)MROWS * 96 * 2);
    float*          buf_dl  = (float*)         alloc((size_t)MROWS * DI * 4);
    unsigned short* buf_y   = (unsigned short*)alloc((size_t)MROWS * DI * 2);
    float*          buf_xo  = (float*)         alloc((size_t)MROWS * DIM * 4);
    float*          buf_hsc = (float*)         alloc((size_t)BB * CHUNK * DSTATE * DI * 4);
    float*          buf_sdt = (float*)         alloc((size_t)BB * CHUNK * DI * 4);

    // ---- one-time f32 -> bf16 conversions ----
    auto cvt = [&](const float* s, unsigned short* d, size_t n) {
        int blocks = (int)((n + 255) / 256); if (blocks > 4096) blocks = 4096;
        k_cvt<<<blocks, 256, 0, stream>>>(s, d, n);
    };
    cvt(x_in,       buf_x,  (size_t)MROWS * DIM);
    cvt(in_proj_w,  wbf_in, (size_t)DEPTH * 2 * DI * DIM);
    cvt(x_proj_w,   wbf_xp, (size_t)DEPTH * 96 * DI);
    cvt(dt_proj_w,  wbf_dt, (size_t)DEPTH * DI * DTRANK);
    cvt(out_proj_w, wbf_op, (size_t)DEPTH * DIM * DI);

    const dim3 blk(256);
    for (int i = 0; i < DEPTH; ++i) {
        // 1) in_proj: [4096,1024] x [4096,1024]^T -> xz [4096,4096] bf16
        k_gemm<0, unsigned short><<<dim3(2 * DI / 128, MROWS / 128), blk, 0, stream>>>(
            buf_x, DIM, wbf_in + (size_t)i * 2 * DI * DIM, DIM,
            buf_xz, 2 * DI, MROWS, 2 * DI, DIM, nullptr);

        // 2) depthwise conv + SiLU -> xs_act bf16
        k_conv<<<(MROWS * DI) / 256, blk, 0, stream>>>(
            buf_xz, conv_w + (size_t)i * DI * DCONV, conv_b + (size_t)i * DI, buf_xs);

        // 3) x_proj: [4096,2048] x [96,2048]^T -> xdbl [4096,96] bf16
        k_gemm<0, unsigned short><<<dim3(1, MROWS / 128), blk, 0, stream>>>(
            buf_xs, DI, wbf_xp + (size_t)i * 96 * DI, DI,
            buf_xdbl, 96, MROWS, 96, DI, nullptr);

        // 4) dt_proj + bias + softplus: [4096,64(ld96)] x [2048,64]^T -> delta f32
        k_gemm<1, float><<<dim3(DI / 128, MROWS / 128), blk, 0, stream>>>(
            buf_xdbl, 96, wbf_dt + (size_t)i * DI * DTRANK, DTRANK,
            buf_dl, DI, MROWS, DI, DTRANK, dt_proj_b + (size_t)i * DI);

        // 5) chunk-parallel selective scan + D skip + silu(z) gating -> y bf16
        const float* Alog_i = A_log + (size_t)i * DI * DSTATE;
        k_scan1<<<BB * CHUNK * (DI / 256), blk, 0, stream>>>(
            buf_xs, buf_dl, buf_xdbl, Alog_i, buf_hsc, buf_sdt);
        k_scan2<<<BB * (DI / 256), blk, 0, stream>>>(Alog_i, buf_hsc, buf_sdt);
        k_scan3<<<BB * CHUNK * (DI / 256), blk, 0, stream>>>(
            buf_xs, buf_dl, buf_xdbl, buf_xz, Alog_i, Dp + (size_t)i * DI,
            buf_hsc, buf_y);

        // 6) out_proj: [4096,2048] x [1024,2048]^T -> xout f32
        k_gemm<0, float><<<dim3(DIM / 128, MROWS / 128), blk, 0, stream>>>(
            buf_y, DI, wbf_op + (size_t)i * DIM * DI, DI,
            buf_xo, DIM, MROWS, DIM, DI, nullptr);

        // 7) RMSNorm -> next layer input (bf16), or final f32 output
        if (i < DEPTH - 1)
            k_rms<unsigned short><<<MROWS, blk, 0, stream>>>(
                buf_xo, rms_w + (size_t)i * DIM, buf_x);
        else
            k_rms<float><<<MROWS, blk, 0, stream>>>(
                buf_xo, rms_w + (size_t)i * DIM, (float*)d_out);
    }
}
